// SemiDANSEplus_62955630625311
// MI455X (gfx1250) — matrix operand order repaired and checked by reading, not compiled, not run on hardware
//
#include <hip/hip_runtime.h>
#include <hip/hip_bf16.h>
#include <math.h>

// ---------------------------------------------------------------------------
// SemiDANSE+ ELBO on gfx1250 (MI455X).
//   N_STATES=3, N_OBS=3, HID=64, N_MC=128, BATCH=64, T=500
//
// Kernels (stream-ordered):
//   k_init    : d_out[0] = constant term of the ELBO
//   k_prep    : pack [Wh | W_mu | W_var | pad] -> f16, transposed to the
//               WMMA B-operand layout (lane = output column, K contiguous)
//   k_cwinv   : per-batch 3x3 inverse + logdet of Cw; logdet folded into d_out
//   k_gru     : 500-step GRU scan, one workgroup of 13 waves (416 threads).
//               Per step: gh(64x208) = h(64x64) @ Waug(64x208) via
//               v_wmma_f32_16x16x32_f16 (8 WMMAs/wave), then elementwise
//               GRU gate update in f32; writes mu and sd=sqrt(softplus) to ws.
//   k_mc      : Monte-Carlo quadratic form, block per (l,n), HBM-stream of
//               eps (49 MB), reduced via wave shuffles + atomicAdd.
// ---------------------------------------------------------------------------

typedef __attribute__((ext_vector_type(16))) _Float16 v16h;
typedef __attribute__((ext_vector_type(8)))  _Float16 v8h;
typedef __attribute__((ext_vector_type(8)))  float    v8f;

#define HIDN   64
#define NB     64
#define TLEN   500
#define NOBS   3
#define NMC    128
#define NCOL   208            // 192 gates + 3 mu + 3 var + 10 pad = 13 tiles
#define NWAVE  13
#define GRUTHREADS (NWAVE * 32)

// ------------------------------- init --------------------------------------
__global__ void k_init(float* out) {
    // const/(T*o) = -0.5 * d * L * log(2*pi) / o  (d==o==3 -> -64*log(2pi))
    out[0] = -0.5f * 3.0f * 500.0f * 128.0f * logf(6.283185307179586f)
             / (500.0f * 3.0f);
}

// ------------------------------- prep --------------------------------------
// whT[n][k], n in [0,208), k in [0,64): f16 transposed augmented weights.
__global__ void k_prep(const float* __restrict__ Wh,
                       const float* __restrict__ Wmu,
                       const float* __restrict__ Wvar,
                       _Float16* __restrict__ whT) {
    int i = blockIdx.x * 256 + threadIdx.x;
    if (i >= NCOL * HIDN) return;
    int n = i >> 6;          // output column
    int k = i & 63;          // input (hidden) index
    float v;
    if (n < 192)      v = Wh[k * 192 + n];
    else if (n < 195) v = Wmu[k * 3 + (n - 192)];
    else if (n < 198) v = Wvar[k * 3 + (n - 195)];
    else              v = 0.0f;
    whT[n * HIDN + k] = (_Float16)v;
}

// ------------------------------- cwinv -------------------------------------
__global__ void k_cwinv(const float* __restrict__ Cw,
                        float* __restrict__ cinv,
                        float* out) {
    int n = threadIdx.x;                      // 64 threads
    const float* m = Cw + n * 9;
    float a = m[0], b = m[1], c = m[2];
    float d = m[3], e = m[4], f = m[5];
    float g = m[6], h = m[7], i = m[8];
    float A = e * i - f * h;
    float B = f * g - d * i;
    float C = d * h - e * g;
    float det = a * A + b * B + c * C;
    float inv = 1.0f / det;
    float* ci = cinv + n * 9;
    ci[0] = A * inv;               ci[1] = (c * h - b * i) * inv;  ci[2] = (b * f - c * e) * inv;
    ci[3] = B * inv;               ci[4] = (a * i - c * g) * inv;  ci[5] = (c * d - a * f) * inv;
    ci[6] = C * inv;               ci[7] = (b * g - a * h) * inv;  ci[8] = (a * e - b * d) * inv;
    // elbo += -0.5*logdet[n]/(N*o)
    atomicAdd(out, -0.5f * logf(det) / (64.0f * 3.0f));
}

// ------------------------------- gru ---------------------------------------
__global__ __launch_bounds__(GRUTHREADS, 1)
void k_gru(const float* __restrict__ Yi,
           const float* __restrict__ Wx,
           const float* __restrict__ bx,
           const float* __restrict__ bh,
           const float* __restrict__ bmu,
           const float* __restrict__ bvar,
           const _Float16* __restrict__ whT,
           float* __restrict__ mu_ws,
           float* __restrict__ sd_ws) {
    __shared__ _Float16 h16[NB * HIDN];      // 8 KB  : h state, f16, row-major
    __shared__ float    gh[NB * NCOL];       // 53 KB : h @ Waug (f32 accum)
    __shared__ float    yis[NB * NOBS];      // Yi[:, t, :]
    __shared__ float    bmu_s[3], bvar_s[3];

    const int tid = threadIdx.x;
    const int w   = tid >> 5;                // wave index: output N-column tile
    const int l   = tid & 31;                // lane
    const int l16 = l & 15;
    const int hi16 = (l & 16) ? 1 : 0;

    // ---- init: h0 = 0 ----
    for (int i = tid; i < NB * HIDN; i += GRUTHREADS) h16[i] = (_Float16)0.0f;
    if (tid < 3) { bmu_s[tid] = bmu[tid]; bvar_s[tid] = bvar[tid]; }

    // ---- load this wave's B operand (2 K-tiles) once into registers ----
    // B layout (32x16, f16): lane holds column n = w*16 + (l&15);
    // lanes 0-15 carry K rows [kt*32, kt*32+16), lanes 16-31 carry +16.
    const int ncol = w * 16 + l16;
    const _Float16* bp = whT + ncol * HIDN;
    v16h B0 = *(const v16h*)(bp + (hi16 ? 16 : 0));        // K 0..31
    v16h B1 = *(const v16h*)(bp + 32 + (hi16 ? 16 : 0));   // K 32..63

    __syncthreads();

    for (int t = 0; t < TLEN; ++t) {
        // ---- stage Yi[:, t, :] ----
        for (int i = tid; i < NB * NOBS; i += GRUTHREADS) {
            int b = i / 3, o = i - b * 3;
            yis[i] = Yi[(b * TLEN + t) * 3 + o];
        }

        // ---- gh = h @ Waug via WMMA: 4 M-tiles x (this wave's N-tile) ----
        const int colg  = ncol;
        const int a0    = hi16 ? 16 : 0;         // A-operand K chunk select
#pragma unroll
        for (int mt = 0; mt < 4; ++mt) {
            const _Float16* hrow = &h16[(mt * 16 + l16) * HIDN];
            // A tile, K 0..31  (elems 0-15 -> K a0..a0+15)
            v8h lo0 = *(const v8h*)(hrow + a0);
            v8h hi0 = *(const v8h*)(hrow + a0 + 8);
            v16h A0 = __builtin_shufflevector(lo0, hi0,
                0,1,2,3,4,5,6,7,8,9,10,11,12,13,14,15);
            // A tile, K 32..63
            v8h lo1 = *(const v8h*)(hrow + 32 + a0);
            v8h hi1 = *(const v8h*)(hrow + 40 + a0);
            v16h A1 = __builtin_shufflevector(lo1, hi1,
                0,1,2,3,4,5,6,7,8,9,10,11,12,13,14,15);

            v8f acc = {0.f,0.f,0.f,0.f,0.f,0.f,0.f,0.f};
            acc = __builtin_amdgcn_wmma_f32_16x16x32_f16(
                      false, A0, false, B0, (short)0, acc, false, false);
            acc = __builtin_amdgcn_wmma_f32_16x16x32_f16(
                      false, A1, false, B1, (short)0, acc, false, false);

            // D layout: lane = col (l&15), VGPR e -> row e + 8*(l>>4)
            const int rbase = mt * 16 + 8 * hi16;
#pragma unroll
            for (int e = 0; e < 8; ++e)
                gh[(rbase + e) * NCOL + colg] = acc[e];
        }
        __syncthreads();

        // ---- elementwise GRU gate update ----
        for (int idx = tid; idx < NB * HIDN; idx += GRUTHREADS) {
            int b = idx >> 6, j = idx & 63;
            float y0 = yis[b * 3 + 0], y1 = yis[b * 3 + 1], y2 = yis[b * 3 + 2];
            float xz = bx[j]       + y0 * Wx[j]       + y1 * Wx[192 + j]       + y2 * Wx[384 + j];
            float xr = bx[64 + j]  + y0 * Wx[64 + j]  + y1 * Wx[256 + j]       + y2 * Wx[448 + j];
            float xn = bx[128 + j] + y0 * Wx[128 + j] + y1 * Wx[320 + j]       + y2 * Wx[512 + j];
            float hz = gh[b * NCOL + j]        + bh[j];
            float hr = gh[b * NCOL + 64 + j]   + bh[64 + j];
            float hn = gh[b * NCOL + 128 + j]  + bh[128 + j];
            float z  = 1.0f / (1.0f + __expf(-(xz + hz)));
            float r  = 1.0f / (1.0f + __expf(-(xr + hr)));
            float nn = tanhf(xn + r * hn);
            float ho = (float)h16[idx];
            float hw = (1.0f - z) * nn + z * ho;
            h16[idx] = (_Float16)hw;
        }
        // ---- prior head: mu and sd = sqrt(softplus(.)) ----
        if (tid < NB * 6) {
            int b = tid / 6, c = tid - b * 6;
            float v = gh[b * NCOL + 192 + c];
            if (c < 3) {
                mu_ws[t * (NB * 3) + b * 3 + c] = v + bmu_s[c];
            } else {
                float x  = v + bvar_s[c - 3];
                float sp = (x > 20.0f) ? x : log1pf(__expf(x));
                sd_ws[t * (NB * 3) + b * 3 + (c - 3)] = sqrtf(sp);
            }
        }
        __syncthreads();
    }
}

// ------------------------------- mc ----------------------------------------
__global__ __launch_bounds__(256)
void k_mc(const float* __restrict__ eps,
          const float* __restrict__ Yi,
          const float* __restrict__ H,
          const float* __restrict__ cinv,
          const float* __restrict__ mu,
          const float* __restrict__ sd,
          float* out) {
    const int ln = blockIdx.x;               // 0 .. L*N-1
    const int l  = ln >> 6;
    const int n  = ln & 63;

    float h00 = H[0], h01 = H[1], h02 = H[2];
    float h10 = H[3], h11 = H[4], h12 = H[5];
    float h20 = H[6], h21 = H[7], h22 = H[8];
    const float* ci = cinv + n * 9;
    float c00 = ci[0], c01 = ci[1], c02 = ci[2];
    float c10 = ci[3], c11 = ci[4], c12 = ci[5];
    float c20 = ci[6], c21 = ci[7], c22 = ci[8];

    float q = 0.0f;
    for (int t = threadIdx.x; t < TLEN; t += 256) {
        const float* e = eps + ((size_t)ln * TLEN + t) * 3;
        const float* m = mu + t * (NB * 3) + n * 3;
        const float* s = sd + t * (NB * 3) + n * 3;
        float x0 = m[0] + s[0] * e[0];
        float x1 = m[1] + s[1] * e[1];
        float x2 = m[2] + s[2] * e[2];
        const float* y = Yi + ((size_t)n * TLEN + t) * 3;
        float d0 = y[0] - (h00 * x0 + h01 * x1 + h02 * x2);
        float d1 = y[1] - (h10 * x0 + h11 * x1 + h12 * x2);
        float d2 = y[2] - (h20 * x0 + h21 * x1 + h22 * x2);
        q += d0 * (c00 * d0 + c01 * d1 + c02 * d2)
           + d1 * (c10 * d0 + c11 * d1 + c12 * d2)
           + d2 * (c20 * d0 + c21 * d1 + c22 * d2);
    }
    // wave32 shuffle reduction, then cross-wave via LDS
    for (int off = 16; off > 0; off >>= 1) q += __shfl_down(q, off, 32);
    __shared__ float red[8];
    if ((threadIdx.x & 31) == 0) red[threadIdx.x >> 5] = q;
    __syncthreads();
    if (threadIdx.x == 0) {
        float s = 0.0f;
#pragma unroll
        for (int i = 0; i < 8; ++i) s += red[i];
        // elbo += -0.5 * quad / (N * L * T * o)
        atomicAdd(out, s * (-0.5f / (64.0f * 128.0f * 500.0f * 3.0f)));
    }
}

// ------------------------------- launch ------------------------------------
extern "C" void kernel_launch(void* const* d_in, const int* in_sizes, int n_in,
                              void* d_out, int out_size, void* d_ws, size_t ws_size,
                              hipStream_t stream) {
    const float* Yi   = (const float*)d_in[0];
    const float* Cw   = (const float*)d_in[1];
    const float* eps  = (const float*)d_in[2];
    const float* H    = (const float*)d_in[3];
    const float* Wx   = (const float*)d_in[4];
    const float* Wh   = (const float*)d_in[5];
    const float* bx   = (const float*)d_in[6];
    const float* bh   = (const float*)d_in[7];
    const float* Wmu  = (const float*)d_in[8];
    const float* bmu  = (const float*)d_in[9];
    const float* Wvar = (const float*)d_in[10];
    const float* bvar = (const float*)d_in[11];
    float* out = (float*)d_out;

    // workspace layout (bytes, 32B-aligned regions):
    //   [0, 26624)        whT  : f16 [208][64]
    //   [26624, 28928)    cinv : f32 [64][9]
    //   [28928, 412928)   mu   : f32 [500][64][3]
    //   [412928, 796928)  sd   : f32 [500][64][3]
    char* ws = (char*)d_ws;
    _Float16* whT = (_Float16*)ws;
    float* cinv   = (float*)(ws + 26624);
    float* mu     = (float*)(ws + 28928);
    float* sd     = (float*)(ws + 412928);

    k_init<<<1, 1, 0, stream>>>(out);
    k_prep<<<(NCOL * HIDN + 255) / 256, 256, 0, stream>>>(Wh, Wmu, Wvar, whT);
    k_cwinv<<<1, 64, 0, stream>>>(Cw, cinv, out);
    k_gru<<<1, GRUTHREADS, 0, stream>>>(Yi, Wx, bx, bh, bmu, bvar, whT, mu, sd);
    k_mc<<<NMC * NB, 256, 0, stream>>>(eps, Yi, H, cinv, mu, sd, out);
}
